// GSchNet_45767171506192
// MI455X (gfx1250) — matrix operand order repaired and checked by reading, not compiled, not run on hardware
//
#include <hip/hip_runtime.h>
#include <math.h>

// ---------------------------------------------------------------------------
// GSchNet forward for MI455X (gfx1250, wave32, WMMA).
// All 128-wide GEMMs on v_wmma_f32_16x16x32_bf16 (K=128 -> 4 WMMA steps).
// Activations kept bf16 end-to-end: LDS tiles are bf16 so A fragments are two
// contiguous ds_load_b128 with no conversion VALU; h lives in bf16 in global
// workspace, halving the 400k-edge random gather traffic. Weights converted
// once to bf16 column-major so each lane's B fragment is one 32-byte load.
// Node-MLP tile staging uses async global->LDS copies when available.
// Shifted softplus uses hardware transcendentals (v_exp_f32/v_log_f32).
// Only the LAST interaction's edge MLP is computed (reference overwrites
// `edges` each iteration and never uses intermediates).
// ---------------------------------------------------------------------------

#define LAT 128
#define NRBF 100
#define NELEM 5
#define NRADII 64

typedef __attribute__((ext_vector_type(16))) __bf16 v16bf;
typedef __attribute__((ext_vector_type(8)))  __bf16 v8bf;
typedef __attribute__((ext_vector_type(8)))  float  v8f;
typedef __attribute__((ext_vector_type(4)))  int    v4i;

#if __has_builtin(__builtin_amdgcn_global_load_async_to_lds_b128)
#define HAVE_ASYNC_LDS 1
#else
#define HAVE_ASYNC_LDS 0
#endif

__device__ __forceinline__ float sspf(float x) {
  // softplus(x) - log(2) = max(x,0) + log(1 + exp(-|x|)) - log(2)
  // fast-math transcendentals: one v_exp_f32 + one v_log_f32
  return fmaxf(x, 0.0f) + __logf(1.0f + __expf(-fabsf(x))) - 0.69314718055994531f;
}

__device__ __forceinline__ __bf16 f2bf(float f) {
  union { float f; unsigned int u; } in; in.f = f;
  unsigned int r = in.u + 0x7FFFu + ((in.u >> 16) & 1u);  // RNE
  union { unsigned short s; __bf16 b; } out;
  out.s = (unsigned short)(r >> 16);
  return out.b;
}

__device__ __forceinline__ float bf2f(__bf16 b) {
  union { unsigned short s; __bf16 b; } in; in.b = b;
  union { unsigned int u; float f; } out; out.u = ((unsigned int)in.s) << 16;
  return out.f;
}
// unpack packed bf16 pair from a u32 word
__device__ __forceinline__ float bflo(unsigned int w) {
  union { unsigned int u; float f; } o; o.u = w << 16; return o.f;
}
__device__ __forceinline__ float bfhi(unsigned int w) {
  union { unsigned int u; float f; } o; o.u = w & 0xFFFF0000u; return o.f;
}

// A fragment (16x32 bf16) from a bf16 LDS tile (16 rows x 128, row-major):
// two contiguous 16B loads per lane (lanes 0-15 K=0..15, lanes 16-31 K=16..31).
__device__ __forceinline__ v16bf load_a_frag(const __bf16* tile, int ks, int lane) {
  const int m  = lane & 15;
  const int kh = (lane >> 4) * 16;
  const __bf16* p = tile + m * LAT + ks * 32 + kh;
  union { v16bf v; v8bf h[2]; } u;
  u.h[0] = *(const v8bf*)p;
  u.h[1] = *(const v8bf*)(p + 8);
  return u.v;
}

// B fragment (32x16 bf16) from bf16 column-major weights Wcm[N][128]:
// lane n = lane&15 is column; lanes 0-15 hold K=kb..kb+15 contiguously.
__device__ __forceinline__ v16bf load_b_frag(const __bf16* Wcm, int ncb, int ks, int lane) {
  const int n  = lane & 15;
  const int kb = ks * 32 + (lane >> 4) * 16;
  return *(const v16bf*)(Wcm + (size_t)(ncb + n) * LAT + kb);
}

// 16x16 output tile of tile(16x128 bf16) @ Wcm, K=128.
__device__ __forceinline__ v8f gemm_tile(const __bf16* tile, const __bf16* Wcm, int ncb, int lane) {
  v8f c = {};
#pragma unroll
  for (int ks = 0; ks < 4; ++ks) {
    v16bf a = load_a_frag(tile, ks, lane);
    v16bf b = load_b_frag(Wcm, ncb, ks, lane);
    c = __builtin_amdgcn_wmma_f32_16x16x32_bf16(false, a, false, b, (short)0, c,
                                                false, false);
  }
  return c;
}

// Store 16x16 f32 D tile into a bf16 LDS tile with bias (+optional ssp).
// C/D layout: lane n = lane&15; VGPR j holds row m = j + 8*(lane>>4).
__device__ __forceinline__ void store_tile(__bf16* out, v8f c, const float* bias,
                                           int ncb, int lane, bool act) {
  const int n  = ncb + (lane & 15);
  const int mb = (lane >> 4) * 8;
  const float bv = bias[n];
#pragma unroll
  for (int j = 0; j < 8; ++j) {
    float v = c[j] + bv;
    if (act) v = sspf(v);
    out[(mb + j) * LAT + n] = f2bf(v);
  }
}

// Stage 16 rows x 128 of bf16 h into LDS (async fast path for full tiles).
__device__ __forceinline__ void stage_h_tile(__bf16* tile, const __bf16* hin,
                                             int row0, int nrows) {
#if HAVE_ASYNC_LDS
  if (row0 + 16 <= nrows) {
    const __bf16* gsrc = hin + (size_t)row0 * LAT + threadIdx.x * 8;  // 16B/thread
    __bf16* lds = tile + threadIdx.x * 8;
    __builtin_amdgcn_global_load_async_to_lds_b128(
        (__attribute__((address_space(1))) v4i*)gsrc,
        (__attribute__((address_space(3))) v4i*)lds, 0, 0);
#if __has_builtin(__builtin_amdgcn_s_wait_asynccnt)
    __builtin_amdgcn_s_wait_asynccnt(0);
#else
    asm volatile("s_wait_asynccnt 0" ::: "memory");
#endif
    return;
  }
#endif
  {
    int elem = threadIdx.x * 8;  // 8 bf16 = 16 bytes per thread
    int r = row0 + (elem >> 7);
    uint4 v = {0u, 0u, 0u, 0u};
    if (r < nrows) v = *(const uint4*)(hin + (size_t)row0 * LAT + elem);
    *(uint4*)((char*)tile + threadIdx.x * 16) = v;
  }
}

// ------------------------- small utility kernels ---------------------------

// f32 row-major W[K][N] -> bf16 column-major Wcm[N][Kp] (zero pad K..Kp).
__global__ void cvt_cm_kernel(const float* __restrict__ W, __bf16* __restrict__ Wcm,
                              int K, int N, int Kp) {
  int i = blockIdx.x * blockDim.x + threadIdx.x;
  if (i >= N * Kp) return;
  int n = i / Kp, k = i - n * Kp;
  Wcm[(size_t)n * Kp + k] = f2bf((k < K) ? W[(size_t)k * N + n] : 0.0f);
}

__global__ void embed_kernel(const float* __restrict__ embed, const int* __restrict__ species,
                             __bf16* __restrict__ h, int N) {
  int i = blockIdx.x * blockDim.x + threadIdx.x;
  if (i >= N * LAT) return;
  int n = i >> 7, c = i & 127;
  h[i] = f2bf(embed[species[n] * LAT + c]);
}

// node->graph map via prefix sum of n_node; also zero the species-logit output.
__global__ void seg_zero_kernel(const int* __restrict__ n_node, int* __restrict__ n2g,
                                float* __restrict__ logits, int G, int N) {
  __shared__ int off[256];
  if (threadIdx.x == 0) {
    int acc = 0;
    for (int g = 0; g < G; ++g) { off[g] = acc; acc += n_node[g]; }
  }
  __syncthreads();
  for (int n = threadIdx.x; n < N; n += blockDim.x) {
    int lo = 0, hi = G - 1;
    while (lo < hi) { int mid = (lo + hi + 1) >> 1; if (off[mid] <= n) lo = mid; else hi = mid - 1; }
    n2g[n] = lo;
  }
  for (int i = threadIdx.x; i < G * NELEM; i += blockDim.x) logits[i] = 0.0f;
}

// ------------------------------ node MLP -----------------------------------
__global__ __launch_bounds__(256) void node_mlp_kernel(
    const __bf16* __restrict__ hin, __bf16* __restrict__ hout,
    const __bf16* __restrict__ W0, const float* __restrict__ b0,
    const __bf16* __restrict__ W1, const float* __restrict__ b1,
    const __bf16* __restrict__ W2, const float* __restrict__ b2, int nrows) {
  __shared__ alignas(16) __bf16 tA[16 * LAT];
  __shared__ alignas(16) __bf16 tB[16 * LAT];
  const int row0 = blockIdx.x * 16;
  const int lane = threadIdx.x & 31;
  const int wave = threadIdx.x >> 5;
  const int ncb  = wave * 16;
  __builtin_prefetch(W0, 0, 1);

  stage_h_tile(tA, hin, row0, nrows);
  __syncthreads();

  v8f c = gemm_tile(tA, W0, ncb, lane);
  store_tile(tB, c, b0, ncb, lane, true);
  __syncthreads();
  c = gemm_tile(tB, W1, ncb, lane);
  store_tile(tA, c, b1, ncb, lane, true);
  __syncthreads();
  c = gemm_tile(tA, W2, ncb, lane);
  {
    const int n  = ncb + (lane & 15);
    const int mb = (lane >> 4) * 8;
    const float bv = b2[n];
#pragma unroll
    for (int j = 0; j < 8; ++j) {
      int r = row0 + mb + j;
      if (r < nrows) hout[(size_t)r * LAT + n] = f2bf(c[j] + bv);  // no act on final layer
    }
  }
}

// ------------------------------ edge kernel --------------------------------
// Per 16 edges: d = ||h[s]-h[r]||, RBF(100, zero-padded to 128),
// then 2-layer MLP (100->128 ssp, 128->128) on WMMA. Writes f32 edges output.
__global__ __launch_bounds__(256) void edge_kernel(
    const __bf16* __restrict__ h, const int* __restrict__ senders,
    const int* __restrict__ receivers,
    const __bf16* __restrict__ WrCM, const float* __restrict__ br,
    const __bf16* __restrict__ W2CM, const float* __restrict__ b2,
    float* __restrict__ out, int nedges) {
  __shared__ alignas(16) __bf16 tA[16 * LAT];
  __shared__ alignas(16) __bf16 tB[16 * LAT];
  __shared__ float dsh[16];
  const int e0   = blockIdx.x * 16;
  const int lane = threadIdx.x & 31;
  const int wave = threadIdx.x >> 5;
  const int ncb  = wave * 16;

  {  // distances: 16 threads/edge, each decodes one 16B chunk (8 bf16)
    const int el = threadIdx.x >> 4;
    const int part = threadIdx.x & 15;
    const int e = e0 + el;
    float acc = 0.0f;
    if (e < nedges) {
      uint4 ua = *((const uint4*)(h + (size_t)senders[e] * LAT) + part);
      uint4 ub = *((const uint4*)(h + (size_t)receivers[e] * LAT) + part);
      unsigned int wa[4] = {ua.x, ua.y, ua.z, ua.w};
      unsigned int wb[4] = {ub.x, ub.y, ub.z, ub.w};
#pragma unroll
      for (int i = 0; i < 4; ++i) {
        float d0 = bflo(wa[i]) - bflo(wb[i]);
        float d1 = bfhi(wa[i]) - bfhi(wb[i]);
        acc += d0 * d0 + d1 * d1;
      }
    }
#pragma unroll
    for (int o = 8; o; o >>= 1) acc += __shfl_xor(acc, o, 32);
    if (part == 0) dsh[el] = sqrtf(acc);
  }
  __syncthreads();

  for (int i = threadIdx.x; i < 16 * LAT; i += 256) {  // RBF tile, K-padded
    int el = i >> 7, k = i & 127;
    float v = 0.0f;
    if (k < NRBF) {
      float t = dsh[el] - (float)k * (10.0f / 99.0f);
      v = __expf(-t * t);  // GAMMA = 1
    }
    tA[i] = f2bf(v);
  }
  __syncthreads();

  v8f c = gemm_tile(tA, WrCM, ncb, lane);
  store_tile(tB, c, br, ncb, lane, true);
  __syncthreads();
  c = gemm_tile(tB, W2CM, ncb, lane);
  {
    const int n  = ncb + (lane & 15);
    const int mb = (lane >> 4) * 8;
    const float bv = b2[n];
#pragma unroll
    for (int j = 0; j < 8; ++j) {
      int e = e0 + mb + j;
      if (e < nedges) out[(size_t)e * LAT + n] = c[j] + bv;
    }
  }
}

// ---------------------------- species head ---------------------------------
__global__ __launch_bounds__(256) void species_kernel(
    const __bf16* __restrict__ h, const float* __restrict__ embed,
    const __bf16* __restrict__ W0, const float* __restrict__ b0,
    const __bf16* __restrict__ W1, const float* __restrict__ b1,
    const float* __restrict__ W2, const float* __restrict__ b2,
    const int* __restrict__ n2g, float* __restrict__ logits, int nrowsTot) {
  __shared__ alignas(16) __bf16 tA[16 * LAT];
  __shared__ alignas(16) __bf16 tB[16 * LAT];
  const int row0 = blockIdx.x * 16;
  const int lane = threadIdx.x & 31;
  const int wave = threadIdx.x >> 5;
  const int ncb  = wave * 16;

  for (int i = threadIdx.x; i < 16 * LAT; i += 256) {
    int r = row0 + (i >> 7), cch = i & 127;
    float v = 0.0f;
    if (r < nrowsTot) {
      int n = r / 5, s = r - n * 5;
      v = bf2f(h[(size_t)n * LAT + cch]) * embed[s * LAT + cch];
    }
    tA[i] = f2bf(v);
  }
  __syncthreads();

  v8f c = gemm_tile(tA, W0, ncb, lane);
  store_tile(tB, c, b0, ncb, lane, true);
  __syncthreads();
  c = gemm_tile(tB, W1, ncb, lane);
  store_tile(tA, c, b1, ncb, lane, true);
  __syncthreads();

  {  // diagonal of 128->5 layer; 16 threads per row; atomic segment sum
    const int m = threadIdx.x >> 4;
    const int part = threadIdx.x & 15;
    const int r = row0 + m;
    const int sidx = r % 5;
    float sum = 0.0f;
    for (int k = part; k < LAT; k += 16) sum += bf2f(tA[m * LAT + k]) * W2[k * 5 + sidx];
#pragma unroll
    for (int o = 8; o; o >>= 1) sum += __shfl_xor(sum, o, 32);
    if (part == 0 && r < nrowsTot) {
      int n = r / 5;
      atomicAdd(&logits[n2g[n] * NELEM + sidx], sum + b2[sidx]);
    }
  }
}

// ----------------------------- distance head -------------------------------
__global__ __launch_bounds__(256) void dist_kernel(
    const __bf16* __restrict__ h, const float* __restrict__ embed,
    const int* __restrict__ tgt,
    const __bf16* __restrict__ W0, const float* __restrict__ b0,
    const __bf16* __restrict__ W1, const float* __restrict__ b1,
    const __bf16* __restrict__ W2, const float* __restrict__ b2,
    float* __restrict__ out, int nrows) {
  __shared__ alignas(16) __bf16 tA[16 * LAT];
  __shared__ alignas(16) __bf16 tB[16 * LAT];
  const int row0 = blockIdx.x * 16;
  const int lane = threadIdx.x & 31;
  const int wave = threadIdx.x >> 5;
  const int ncb  = wave * 16;
  const float* e = embed + (size_t)tgt[0] * LAT;

  for (int i = threadIdx.x; i < 16 * LAT; i += 256) {
    int r = row0 + (i >> 7), c = i & 127;
    float v = 0.0f;
    if (r < nrows) v = bf2f(h[(size_t)r * LAT + c]) * e[c];
    tA[i] = f2bf(v);
  }
  __syncthreads();

  v8f c = gemm_tile(tA, W0, ncb, lane);
  store_tile(tB, c, b0, ncb, lane, true);
  __syncthreads();
  c = gemm_tile(tB, W1, ncb, lane);
  store_tile(tA, c, b1, ncb, lane, true);
  __syncthreads();

  if (wave < 4) {  // only 64 output cols
    c = gemm_tile(tA, W2, ncb, lane);
    const int n  = ncb + (lane & 15);
    const int mb = (lane >> 4) * 8;
    const float bv = b2[n];
#pragma unroll
    for (int j = 0; j < 8; ++j) {
      int r = row0 + mb + j;
      if (r < nrows) out[(size_t)r * NRADII + n] = c[j] + bv;
    }
  }
}

// ------------------------------- host side ---------------------------------
// Assumed d_in ordering: JAX pytree flattening of setup_inputs(): params
// leaves first (dict keys sorted: dist_head, edge, embed, node, species_head;
// lists in order; (W,b) tuples in order), then species, senders, receivers,
// n_node, target_species.
extern "C" void kernel_launch(void* const* d_in, const int* in_sizes, int n_in,
                              void* d_out, int out_size, void* d_ws, size_t ws_size,
                              hipStream_t stream) {
  (void)in_sizes; (void)n_in; (void)out_size; (void)ws_size;
  const int N = 25000, E = 400000, G = 250;

  const float* distW[3] = {(const float*)d_in[0], (const float*)d_in[2], (const float*)d_in[4]};
  const float* distB[3] = {(const float*)d_in[1], (const float*)d_in[3], (const float*)d_in[5]};
  const float* edgeWr2 = (const float*)d_in[14];  // edge[2] layer0 W (100x128)
  const float* edgeBr2 = (const float*)d_in[15];
  const float* edgeW22 = (const float*)d_in[16];  // edge[2] layer1 W (128x128)
  const float* edgeB22 = (const float*)d_in[17];
  const float* embed   = (const float*)d_in[18];
  const float* nodeW[3][3]; const float* nodeB[3][3];
  for (int i = 0; i < 3; ++i)
    for (int l = 0; l < 3; ++l) {
      nodeW[i][l] = (const float*)d_in[19 + 6 * i + 2 * l];
      nodeB[i][l] = (const float*)d_in[20 + 6 * i + 2 * l];
    }
  const float* spW[3] = {(const float*)d_in[37], (const float*)d_in[39], (const float*)d_in[41]};
  const float* spB[3] = {(const float*)d_in[38], (const float*)d_in[40], (const float*)d_in[42]};
  const int* species   = (const int*)d_in[43];
  const int* senders   = (const int*)d_in[44];
  const int* receivers = (const int*)d_in[45];
  const int* n_node    = (const int*)d_in[46];
  const int* tgt       = (const int*)d_in[47];

  // workspace: hA, hB (bf16 N x 128), n2g, then bf16 weight slots
  __bf16* hA = (__bf16*)d_ws;
  __bf16* hB = hA + (size_t)N * LAT;
  int* n2g   = (int*)(hB + (size_t)N * LAT);
  size_t woff = ((size_t)(2 * (size_t)N * LAT * 2 + (size_t)N * 4) + 255) & ~(size_t)255;
  __bf16* wbase = (__bf16*)((char*)d_ws + woff);
  const size_t SLOT = (size_t)LAT * LAT;
#define WSLOT(i) (wbase + (size_t)(i) * SLOT)

  float* out_logits = (float*)d_out;                   // G*5
  float* out_dl     = out_logits + (size_t)G * NELEM;  // N*64
  float* out_edges  = out_dl + (size_t)N * NRADII;     // E*128

  const int cvtBlocks = (LAT * LAT + 255) / 256;
  for (int i = 0; i < 3; ++i)   // node weights -> slots 0..8
    for (int l = 0; l < 3; ++l)
      cvt_cm_kernel<<<cvtBlocks, 256, 0, stream>>>(nodeW[i][l], WSLOT(3 * i + l), LAT, LAT, LAT);
  cvt_cm_kernel<<<cvtBlocks, 256, 0, stream>>>(edgeWr2, WSLOT(9), NRBF, LAT, LAT);
  cvt_cm_kernel<<<cvtBlocks, 256, 0, stream>>>(edgeW22, WSLOT(10), LAT, LAT, LAT);
  cvt_cm_kernel<<<cvtBlocks, 256, 0, stream>>>(spW[0], WSLOT(11), LAT, LAT, LAT);
  cvt_cm_kernel<<<cvtBlocks, 256, 0, stream>>>(spW[1], WSLOT(12), LAT, LAT, LAT);
  cvt_cm_kernel<<<cvtBlocks, 256, 0, stream>>>(distW[0], WSLOT(13), LAT, LAT, LAT);
  cvt_cm_kernel<<<cvtBlocks, 256, 0, stream>>>(distW[1], WSLOT(14), LAT, LAT, LAT);
  cvt_cm_kernel<<<(LAT * NRADII + 255) / 256, 256, 0, stream>>>(distW[2], WSLOT(15), LAT, NRADII, LAT);

  seg_zero_kernel<<<1, 256, 0, stream>>>(n_node, n2g, out_logits, G, N);
  embed_kernel<<<(N * LAT + 255) / 256, 256, 0, stream>>>(embed, species, hA, N);

  const int nodeBlocks = (N + 15) / 16;
  node_mlp_kernel<<<nodeBlocks, 256, 0, stream>>>(hA, hB, WSLOT(0), nodeB[0][0],
      WSLOT(1), nodeB[0][1], WSLOT(2), nodeB[0][2], N);
  node_mlp_kernel<<<nodeBlocks, 256, 0, stream>>>(hB, hA, WSLOT(3), nodeB[1][0],
      WSLOT(4), nodeB[1][1], WSLOT(5), nodeB[1][2], N);
  // final-iteration edges (uses h after 2 node updates)
  edge_kernel<<<(E + 15) / 16, 256, 0, stream>>>(hA, senders, receivers,
      WSLOT(9), edgeBr2, WSLOT(10), edgeB22, out_edges, E);
  // interaction 2: hA -> hB (final h)
  node_mlp_kernel<<<nodeBlocks, 256, 0, stream>>>(hA, hB, WSLOT(6), nodeB[2][0],
      WSLOT(7), nodeB[2][1], WSLOT(8), nodeB[2][2], N);
  species_kernel<<<(N * NELEM + 15) / 16, 256, 0, stream>>>(hB, embed,
      WSLOT(11), spB[0], WSLOT(12), spB[1], spW[2], spB[2], n2g, out_logits, N * NELEM);
  dist_kernel<<<nodeBlocks, 256, 0, stream>>>(hB, embed, tgt,
      WSLOT(13), distB[0], WSLOT(14), distB[1], WSLOT(15), distB[2], out_dl, N);
#undef WSLOT
}
